// RiemannLinearTransform_60129542144389
// MI455X (gfx1250) — matrix operand order repaired and checked by reading, not compiled, not run on hardware
//
#include <hip/hip_runtime.h>
#include <hip/hip_bf16.h>
#include <math.h>

typedef __attribute__((ext_vector_type(8)))  _Float16 v8h;
typedef __attribute__((ext_vector_type(16))) _Float16 v16h;
typedef __attribute__((ext_vector_type(8)))  float    v8f;

#define TWO_PI_F 6.28318530717958647692f

// ---------------- elementwise helpers ----------------

__global__ void k_cvt_f16(const float* __restrict__ src, _Float16* __restrict__ dst, int n) {
    int i = blockIdx.x * blockDim.x + threadIdx.x;
    if (i < n) dst[i] = (_Float16)src[i];
}

// dst (rows x 2*cols) f16:  [ re | im ]  per row
__global__ void k_pack_pair(const float* __restrict__ re, const float* __restrict__ im,
                            _Float16* __restrict__ dst, int rows, int cols) {
    int i = blockIdx.x * blockDim.x + threadIdx.x;
    int total = rows * cols;
    if (i >= total) return;
    int r = i / cols, c = i % cols;
    size_t base = (size_t)r * (2 * cols);
    dst[base + c]        = (_Float16)re[i];
    dst[base + cols + c] = (_Float16)im[i];
}

// irfft basis, stored transposed for the GEMM "BT" operand:
// out[n * 2J + jj]:
//   jj <  J : w(j)/Nfft * cos(2*pi*j*n/Nfft),  w(0)=1, else 2
//   jj >= J : j=jj-J;  j==0 -> 0 (Im(c0) ignored by irfft), else -(2/Nfft)*sin(2*pi*j*n/Nfft)
__global__ void k_gen_basis(_Float16* __restrict__ out, int nRows, int J, int Nfft) {
    int i = blockIdx.x * blockDim.x + threadIdx.x;
    int rowLen = 2 * J;
    int total = nRows * rowLen;
    if (i >= total) return;
    int n  = i / rowLen;
    int jj = i % rowLen;
    bool isSin = (jj >= J);
    int j = isSin ? (jj - J) : jj;
    int ph = (j * n) % Nfft;                      // exact in int
    float ang = (TWO_PI_F / (float)Nfft) * (float)ph;
    float v;
    if (!isSin) v = (((j == 0) ? 1.0f : 2.0f) / (float)Nfft) * cosf(ang);
    else        v = (j == 0) ? 0.0f : (-2.0f / (float)Nfft) * sinf(ang);
    out[i] = (_Float16)v;
}

// ---------------- WMMA GEMM ----------------
// C[M,N] = A(MxK, row-major f16) * BT(NxK, row-major f16)^T
// OUTF16==1: store f16, optionally scaled per output column (colScale[n])
// OUTF16==0: store f32, optionally + bias[n]
// Block = 256 threads = 8 waves; wave tile 16x64 (4 wmma accumulators); block tile 128x64.
// Requires M%128==0, N%64==0, K%32==0.
template <int OUTF16>
__global__ __launch_bounds__(256) void k_gemm_wmma(
    const _Float16* __restrict__ A, const _Float16* __restrict__ BT,
    void* __restrict__ Cout, const float* __restrict__ colScale,
    const float* __restrict__ bias, int M, int N, int K)
{
    const int lane = threadIdx.x & 31;
    const int wave = threadIdx.x >> 5;
    const int m0 = blockIdx.y * 128 + wave * 16;
    const int n0 = blockIdx.x * 64;
    const int l15 = lane & 15;
    const int lhi = lane >> 4;

    // A fragment: lane row = m0+l15,
    // contiguous halves [k0 + 16*lhi .. +15] (matches B)
    const _Float16* aPtr = A + (size_t)(m0 + l15) * K + lhi * 16;

    // B fragment (contiguous k order, matches A): lane col = n0+16t+l15,
    // contiguous halves [k0 + 16*lhi .. +15] of BT row
    const _Float16* bPtr[4];
#pragma unroll
    for (int t = 0; t < 4; ++t)
        bPtr[t] = BT + (size_t)(n0 + 16 * t + l15) * K + lhi * 16;

    v8f acc[4] = {v8f{}, v8f{}, v8f{}, v8f{}};

    for (int k0 = 0; k0 < K; k0 += 32) {
        v8h a0 = *(const v8h*)(aPtr + k0);
        v8h a1 = *(const v8h*)(aPtr + k0 + 8);
        v16h a = __builtin_shufflevector(a0, a1, 0, 1, 2, 3, 4, 5, 6, 7,
                                                 8, 9, 10, 11, 12, 13, 14, 15);
#pragma unroll
        for (int t = 0; t < 4; ++t) {
            v16h b = *(const v16h*)(bPtr[t] + k0);   // 32B contiguous, 32B-aligned
            acc[t] = __builtin_amdgcn_wmma_f32_16x16x32_f16(
                false, a, false, b, (short)0, acc[t], false, false);
        }
    }

    // D layout: (VGPR r, lane L) = C[m0 + 8*lhi + r, n0 + 16t + l15]
    const int rowBase = m0 + lhi * 8;
    float*    cf = (float*)Cout;
    _Float16* ch = (_Float16*)Cout;
#pragma unroll
    for (int t = 0; t < 4; ++t) {
        int n = n0 + 16 * t + l15;
        float sc = 1.0f, bb = 0.0f;
        if (OUTF16) { if (colScale) sc = colScale[n]; }
        else        { if (bias)     bb = bias[n]; }
#pragma unroll
        for (int r = 0; r < 8; ++r) {
            size_t idx = (size_t)(rowBase + r) * N + n;
            float v = acc[t][r];
            if (OUTF16) ch[idx] = (_Float16)(v * sc);
            else        cf[idx] = v + bb;
        }
    }
}

// ---------------- launcher ----------------

extern "C" void kernel_launch(void* const* d_in, const int* in_sizes, int n_in,
                              void* d_out, int out_size, void* d_ws, size_t ws_size,
                              hipStream_t stream) {
    const float* x    = (const float*)d_in[0];  // (1024, 4096)
    const float* Ur   = (const float*)d_in[1];  // (4096, 512)
    const float* Ui   = (const float*)d_in[2];  // (4096, 512)
    const float* S    = (const float*)d_in[3];  // (2048,)
    const float* Vtr  = (const float*)d_in[4];  // (2048, 1024)
    const float* Vti  = (const float*)d_in[5];  // (2048, 1024)
    const float* bias = (const float*)d_in[6];  // (4096,)
    float* out = (float*)d_out;                 // (1024, 4096)

    const int M = 1024, IN = 4096, OUT = 4096, KR = 2048, JU = 512, JV = 1024;

    char* ws = (char*)d_ws;
    const size_t MB = 1u << 20;
    _Float16* BvT  = (_Float16*)(ws +  0 * MB);  // 4096x2048 f16 = 16 MB (reused for BuT)
    _Float16* catV = (_Float16*)(ws + 16 * MB);  // 2048x2048 f16 =  8 MB (reused for catU)
    _Float16* Vh   = (_Float16*)(ws + 24 * MB);  // 2048x4096 f16 = 16 MB (reused for Uf)
    _Float16* xh   = (_Float16*)(ws + 40 * MB);  // 1024x4096 f16 =  8 MB
    _Float16* t2h  = (_Float16*)(ws + 48 * MB);  // 1024x2048 f16 =  4 MB
    _Float16* BuT  = BvT;                        // 2048x1024 f16 =  4 MB
    _Float16* catU = catV;                       // 4096x1024 f16 =  8 MB
    _Float16* Uf   = Vh;                         // 4096x2048 f16 = 16 MB

    const int TPB = 256;
    auto blocks = [](long n) { return (unsigned)((n + 255) / 256); };

    // 1) Vh basis (BT layout): rows n=0..IN-1, 2*JV halves each
    k_gen_basis<<<blocks((long)IN * 2 * JV), TPB, 0, stream>>>(BvT, IN, JV, IN);
    // 2) catV = [Vt_real | Vt_imag]  (KR x 2*JV)
    k_pack_pair<<<blocks((long)KR * JV), TPB, 0, stream>>>(Vtr, Vti, catV, KR, JV);
    // 3) Vh (KR x IN) = catV (KR x 2JV) * BvT^T
    k_gemm_wmma<1><<<dim3(IN / 64, KR / 128), TPB, 0, stream>>>(
        catV, BvT, (void*)Vh, nullptr, nullptr, KR, IN, 2 * JV);
    // 4) xh = f16(x)
    k_cvt_f16<<<blocks((long)M * IN), TPB, 0, stream>>>(x, xh, M * IN);
    // 5) t2 (M x KR) = (xh * Vh^T), columns scaled by S, stored f16
    k_gemm_wmma<1><<<dim3(KR / 64, M / 128), TPB, 0, stream>>>(
        xh, Vh, (void*)t2h, S, nullptr, M, KR, IN);
    // 6) U basis (BT layout): rows k=0..KR-1, 2*JU halves each   (reuses BvT space)
    k_gen_basis<<<blocks((long)KR * 2 * JU), TPB, 0, stream>>>(BuT, KR, JU, KR);
    // 7) catU = [U_real | U_imag]  (OUT x 2*JU)                  (reuses catV space)
    k_pack_pair<<<blocks((long)OUT * JU), TPB, 0, stream>>>(Ur, Ui, catU, OUT, JU);
    // 8) U (OUT x KR) = catU * BuT^T                             (reuses Vh space)
    k_gemm_wmma<1><<<dim3(KR / 64, OUT / 128), TPB, 0, stream>>>(
        catU, BuT, (void*)Uf, nullptr, nullptr, OUT, KR, 2 * JU);
    // 9) out (M x OUT) = t2 * U^T + bias, f32
    k_gemm_wmma<0><<<dim3(OUT / 64, M / 128), TPB, 0, stream>>>(
        t2h, Uf, (void*)out, nullptr, bias, M, OUT, KR);
}
